// DiscoDownBlock_81037442940978
// MI455X (gfx1250) — matrix operand order repaired and checked by reading, not compiled, not run on hardware
//
#include <hip/hip_runtime.h>
#include <hip/hip_bf16.h>

// ---------------- problem constants ----------------
#define B_      2
#define CIN     128
#define COUT    256
#define HIN     181
#define WIN     360
#define HOUT    91
#define WOUT    180
#define KB      7          // basis count
#define NNZ     16
#define HW1     (HIN*WIN)          // 65160
#define HW2     (HOUT*WOUT)        // 16380
#define GROUPS  8
#define CG1     (CIN/GROUPS)       // 16
#define CG2     (COUT/GROUPS)      // 32
#define KDIM    (CIN*KB)           // 896 = 28*32
#define TOTP    (B_*HW1)           // 130320 total pixels
#define WT32    6                  // ceil(180/32)

typedef __attribute__((ext_vector_type(16))) __bf16 v16bf;
typedef __attribute__((ext_vector_type(8)))  float  v8f;

// ---------------- helpers ----------------
__device__ __forceinline__ unsigned short f2bf(float f) {
  union { float f; unsigned int u; } x; x.f = f;
  unsigned int u = x.u;
  u += 0x7fffu + ((u >> 16) & 1u);   // round-to-nearest-even
  return (unsigned short)(u >> 16);
}

__device__ __forceinline__ float gelu_f(float x) {
  const float c = 0.7978845608028654f; // sqrt(2/pi)
  float t = c * (x + 0.044715f * x * x * x);
  return 0.5f * x * (1.0f + tanhf(t));
}

__device__ __forceinline__ v8f wmma_bf16(v16bf a, v16bf b, v8f c) {
  return __builtin_amdgcn_wmma_f32_16x16x32_bf16(false, a, false, b, (short)0, c,
                                                 false, false);
}

// B-fragment LDS address for N-tile-pair layout [s][nt][lane][elem].
// B 32x16 layout per subtile: lane = ((k%32)/8%2)*16 + n%16 ; element = k%8 + (k%32)/16*8.
__device__ __forceinline__ int bfrag_addr2(int k, int n) {
  int s  = k >> 5;
  int rr = k & 31;
  int nt = n >> 4;
  int nn = n & 15;
  return (((s * 2 + nt) * 32) + (((rr >> 3) & 1) << 4) + nn) * 16 + (rr & 7) + ((rr >> 4) << 3);
}

// ---------------- kernel 0: pack weights into bf16 WMMA A-fragment order ----
// dst[((mt*S+s)*32+l)*16+i] = bf16( src[m*Kdim + k] )
//   m = mt*16 + l%16 ;  k = s*32 + (i%8) + (l/16)*8 + (i/8)*16
__global__ void pack_frag_kernel(const float* __restrict__ src,
                                 unsigned short* __restrict__ dst,
                                 int Kdim, int S, int total) {
  int id = blockIdx.x * blockDim.x + threadIdx.x;
  if (id >= total) return;
  int i  = id & 15;
  int l  = (id >> 4) & 31;
  int sm = id >> 9;
  int s  = sm % S;
  int mt = sm / S;
  int m  = mt * 16 + (l & 15);
  int k  = s * 32 + (i & 7) + ((l >> 4) << 3) + ((i >> 3) << 4);
  dst[id] = f2bf(src[(long)m * Kdim + k]);
}

// ---------------- GroupNorm stats (deterministic two-stage) -----------------
__global__ __launch_bounds__(256)
void gn_partial_kernel(const float* __restrict__ src, float* __restrict__ partials,
                       int C, int Cg, int HW, int chunk) {
  int bg   = blockIdx.x >> 6;    // (b*8+g), 16 total
  int part = blockIdx.x & 63;
  int b = bg >> 3, g = bg & 7;
  long E  = (long)Cg * HW;
  long e0 = (long)part * chunk;
  long e1 = e0 + chunk; if (e1 > E) e1 = E;
  float s = 0.f, q = 0.f;
  for (long e = e0 + threadIdx.x; e < e1; e += 256) {
    int cl  = (int)(e / HW);
    int pix = (int)(e - (long)cl * HW);
    float v = src[((long)(b * C + g * Cg + cl)) * HW + pix];
    s += v; q += v * v;
  }
  __shared__ float ss[256], qq[256];
  ss[threadIdx.x] = s; qq[threadIdx.x] = q;
  __syncthreads();
  for (int o = 128; o > 0; o >>= 1) {
    if (threadIdx.x < o) { ss[threadIdx.x] += ss[threadIdx.x + o];
                           qq[threadIdx.x] += qq[threadIdx.x + o]; }
    __syncthreads();
  }
  if (threadIdx.x == 0) {
    partials[blockIdx.x * 2 + 0] = ss[0];
    partials[blockIdx.x * 2 + 1] = qq[0];
  }
}

__global__ void gn_final_kernel(const float* __restrict__ partials,
                                float* __restrict__ stats, float invE) {
  int bg = blockIdx.x, t = threadIdx.x;        // 16 blocks x 64 threads
  __shared__ float ss[64], qq[64];
  ss[t] = partials[(bg * 64 + t) * 2 + 0];
  qq[t] = partials[(bg * 64 + t) * 2 + 1];
  __syncthreads();
  for (int o = 32; o > 0; o >>= 1) {
    if (t < o) { ss[t] += ss[t + o]; qq[t] += qq[t + o]; }
    __syncthreads();
  }
  if (t == 0) {
    float m = ss[0] * invE;
    float v = qq[0] * invE - m * m;
    stats[bg * 2 + 0] = m;
    stats[bg * 2 + 1] = rsqrtf(v + 1e-5f);
  }
}

// ---------------- fused MLP: GN1 -> w1 -> GELU -> w2 -> +residual -----------
// N-tile = 32 pixels; each A fragment feeds 2 WMMAs (one per N-subtile).
__global__ __launch_bounds__(256)
void mlp_kernel(const float* __restrict__ x, const unsigned short* __restrict__ w1p,
                const float* __restrict__ b1, const unsigned short* __restrict__ w2p,
                const float* __restrict__ b2, const float* __restrict__ gamma,
                const float* __restrict__ beta, const float* __restrict__ stats,
                float* __restrict__ skip) {
  __shared__ __align__(32) unsigned short Bf1[4 * 2 * 32 * 16];  // hn frags (K=128,N=32)
  __shared__ __align__(32) unsigned short Bf2[8 * 2 * 32 * 16];  // act frags (K=256,N=32)
  const int tid = threadIdx.x;
  const long p0 = (long)blockIdx.x * 32;

  // phase 1: normalized activations into B-fragment LDS
  for (int idx = tid; idx < CIN * 32; idx += 256) {
    int k = idx >> 5, n = idx & 31;
    long p = p0 + n;
    if (p >= TOTP) p = TOTP - 1;            // clamp (tail tile), stores are guarded
    int b   = (int)(p / HW1);
    int pix = (int)(p - (long)b * HW1);
    float xv = x[((long)(b * CIN + k)) * HW1 + pix];
    int bg = b * 8 + (k >> 4);
    float hn = (xv - stats[bg * 2]) * stats[bg * 2 + 1] * gamma[k] + beta[k];
    Bf1[bfrag_addr2(k, n)] = f2bf(hn);
  }
  __syncthreads();

  // phase 2: t = GELU(w1 @ hn + b1) -> Bf2
  const int wv = tid >> 5, l = tid & 31, half = l >> 4, ln = l & 15;
  v8f acc00 = {}, acc01 = {}, acc10 = {}, acc11 = {};
  for (int s = 0; s < 4; ++s) {
    v16bf bf0 = *reinterpret_cast<const v16bf*>(&Bf1[((s * 2 + 0) * 32 + l) * 16]);
    v16bf bf1 = *reinterpret_cast<const v16bf*>(&Bf1[((s * 2 + 1) * 32 + l) * 16]);
    v16bf a0 = *reinterpret_cast<const v16bf*>(&w1p[((((2 * wv)     * 4 + s) * 32) + l) * 16]);
    v16bf a1 = *reinterpret_cast<const v16bf*>(&w1p[((((2 * wv + 1) * 4 + s) * 32) + l) * 16]);
    acc00 = wmma_bf16(a0, bf0, acc00);
    acc01 = wmma_bf16(a0, bf1, acc01);
    acc10 = wmma_bf16(a1, bf0, acc10);
    acc11 = wmma_bf16(a1, bf1, acc11);
  }
#pragma unroll
  for (int t = 0; t < 4; ++t) {
    v8f acc = (t == 0) ? acc00 : (t == 1) ? acc01 : (t == 2) ? acc10 : acc11;
    int mt = 2 * wv + (t >> 1);
    int nt = t & 1;
#pragma unroll
    for (int r = 0; r < 8; ++r) {
      int m = mt * 16 + half * 8 + r;
      float v = gelu_f(acc[r] + b1[m]);
      Bf2[bfrag_addr2(m, nt * 16 + ln)] = f2bf(v);
    }
  }
  __syncthreads();

  // phase 3: y = w2 @ t + b2 + x   (wave wv handles M rows [wv*16, wv*16+16))
  v8f accn0 = {}, accn1 = {};
  for (int s = 0; s < 8; ++s) {
    v16bf bf0 = *reinterpret_cast<const v16bf*>(&Bf2[((s * 2 + 0) * 32 + l) * 16]);
    v16bf bf1 = *reinterpret_cast<const v16bf*>(&Bf2[((s * 2 + 1) * 32 + l) * 16]);
    v16bf a = *reinterpret_cast<const v16bf*>(&w2p[(((wv * 8 + s) * 32) + l) * 16]);
    accn0 = wmma_bf16(a, bf0, accn0);
    accn1 = wmma_bf16(a, bf1, accn1);
  }
#pragma unroll
  for (int nt = 0; nt < 2; ++nt) {
    v8f acc = nt ? accn1 : accn0;
    long p = p0 + nt * 16 + ln;
    if (p < TOTP) {
      int b   = (int)(p / HW1);
      int pix = (int)(p - (long)b * HW1);
#pragma unroll
      for (int r = 0; r < 8; ++r) {
        int m = wv * 16 + half * 8 + r;
        long addr = ((long)(b * CIN + m)) * HW1 + pix;
        skip[addr] = acc[r] + b2[m] + x[addr];
      }
    }
  }
}

// ---------------- DISCO: psi-gather -> Y tile -> WMMA channel contraction ---
// N-tile = 32 output longitudes; each wcp A fragment feeds 2 WMMAs.
__global__ __launch_bounds__(256)
void disco_kernel(const float* __restrict__ skip, const float* __restrict__ psi_vals,
                  const int* __restrict__ psi_hi, const int* __restrict__ psi_wi,
                  const unsigned short* __restrict__ wcp, const float* __restrict__ bias,
                  float* __restrict__ outp) {
  __shared__ __align__(32) unsigned short Yf[28 * 2 * 32 * 16];  // 56KB
  __shared__ int   hi_s[NNZ], wi_s[NNZ];
  __shared__ float pv[KB][NNZ];
  const int bx = blockIdx.x;
  const int wt = bx % WT32;
  const int h  = (bx / WT32) % HOUT;
  const int b  = bx / (WT32 * HOUT);
  const int w0 = wt * 32;
  const int tid = threadIdx.x;

  if (tid < NNZ) { hi_s[tid] = psi_hi[h * NNZ + tid]; wi_s[tid] = psi_wi[h * NNZ + tid]; }
  if (tid < KB * NNZ) {
    int k = tid >> 4, n = tid & 15;
    pv[k][n] = psi_vals[((long)k * HOUT + h) * NNZ + n];
  }
  __syncthreads();

  // phase 1: gather + basis reduce, reuse each gathered value for all 7 bases
  for (int idx = tid; idx < CIN * 32; idx += 256) {
    int c = idx >> 5, px = idx & 31;
    int w = w0 + px;
    const float* base = skip + ((long)(b * CIN + c)) * HW1;
    float g[NNZ];
#pragma unroll
    for (int n = 0; n < NNZ; ++n) {
      int col = (wi_s[n] + 2 * w) % WIN;
      g[n] = base[hi_s[n] * WIN + col];
    }
#pragma unroll
    for (int k = 0; k < KB; ++k) {
      float s = 0.f;
#pragma unroll
      for (int n = 0; n < NNZ; ++n) s += pv[k][n] * g[n];
      Yf[bfrag_addr2(c * KB + k, px)] = f2bf(s);
    }
  }
  __syncthreads();

  // phase 2: out[256 x 32] = Wc[256 x 896] @ Y[896 x 32]
  const int wv = tid >> 5, l = tid & 31, half = l >> 4, ln = l & 15;
  const unsigned short* wb0 = wcp + (long)((2 * wv)     * 28) * 32 * 16;
  const unsigned short* wb1 = wcp + (long)((2 * wv + 1) * 28) * 32 * 16;
  v8f acc00 = {}, acc01 = {}, acc10 = {}, acc11 = {};
  for (int s = 0; s < 28; ++s) {
    v16bf bf0 = *reinterpret_cast<const v16bf*>(&Yf[((s * 2 + 0) * 32 + l) * 16]);
    v16bf bf1 = *reinterpret_cast<const v16bf*>(&Yf[((s * 2 + 1) * 32 + l) * 16]);
    v16bf a0 = *reinterpret_cast<const v16bf*>(&wb0[(s * 32 + l) * 16]);
    v16bf a1 = *reinterpret_cast<const v16bf*>(&wb1[(s * 32 + l) * 16]);
    // prefetch slab s+2 of streaming weights (global_prefetch_b8)
    __builtin_prefetch(&wb0[((s + 2) * 32 + l) * 16], 0, 0);
    __builtin_prefetch(&wb1[((s + 2) * 32 + l) * 16], 0, 0);
    acc00 = wmma_bf16(a0, bf0, acc00);
    acc01 = wmma_bf16(a0, bf1, acc01);
    acc10 = wmma_bf16(a1, bf0, acc10);
    acc11 = wmma_bf16(a1, bf1, acc11);
  }
#pragma unroll
  for (int t = 0; t < 4; ++t) {
    v8f acc = (t == 0) ? acc00 : (t == 1) ? acc01 : (t == 2) ? acc10 : acc11;
    int mt = 2 * wv + (t >> 1);
    int nt = t & 1;
    int wcol = w0 + nt * 16 + ln;
    if (wcol < WOUT) {
#pragma unroll
      for (int r = 0; r < 8; ++r) {
        int m = mt * 16 + half * 8 + r;
        outp[(((long)(b * COUT + m)) * HOUT + h) * WOUT + wcol] = acc[r] + bias[m];
      }
    }
  }
}

// ---------------- GN2 apply + GELU --------------------------------------
__global__ void gn2_apply_kernel(const float* __restrict__ src,
                                 const float* __restrict__ stats,
                                 const float* __restrict__ gamma,
                                 const float* __restrict__ beta,
                                 float* __restrict__ dst, long total) {
  long id = (long)blockIdx.x * blockDim.x + threadIdx.x;
  if (id >= total) return;
  long c = (id / HW2) % COUT;
  long b = id / ((long)HW2 * COUT);
  int bg = (int)(b * 8 + (c >> 5));
  float v = (src[id] - stats[bg * 2]) * stats[bg * 2 + 1] * gamma[c] + beta[c];
  dst[id] = gelu_f(v);
}

// ---------------- launcher ----------------
extern "C" void kernel_launch(void* const* d_in, const int* in_sizes, int n_in,
                              void* d_out, int out_size, void* d_ws, size_t ws_size,
                              hipStream_t stream) {
  const float* x        = (const float*)d_in[0];
  const float* psi_vals = (const float*)d_in[1];
  const int*   psi_hi   = (const int*)  d_in[2];
  const int*   psi_wi   = (const int*)  d_in[3];
  const float* weight   = (const float*)d_in[4];
  const float* bias     = (const float*)d_in[5];
  const float* g1       = (const float*)d_in[6];
  const float* be1      = (const float*)d_in[7];
  const float* w1       = (const float*)d_in[8];
  const float* b1       = (const float*)d_in[9];
  const float* w2       = (const float*)d_in[10];
  const float* b2       = (const float*)d_in[11];
  const float* g2       = (const float*)d_in[12];
  const float* be2      = (const float*)d_in[13];

  float* skip  = (float*)d_out;                         // [2,128,181,360]
  float* xdown = (float*)d_out + (long)B_ * CIN * HW1;  // [2,256,91,180]

  // workspace carve-up (64B aligned regions)
  char* ws = (char*)d_ws;
  size_t off = 0;
  auto carve = [&](size_t bytes) { char* p = ws + off; off = (off + bytes + 63) & ~(size_t)63; return p; };
  unsigned short* w1p   = (unsigned short*)carve(16 * 4  * 32 * 16 * 2);   // 64KB
  unsigned short* w2p   = (unsigned short*)carve(8  * 8  * 32 * 16 * 2);   // 64KB
  unsigned short* wcp   = (unsigned short*)carve(16 * 28 * 32 * 16 * 2);   // 448KB
  float* part1  = (float*)carve(16 * 64 * 2 * sizeof(float));
  float* stats1 = (float*)carve(16 * 2 * sizeof(float));
  float* part2  = (float*)carve(16 * 64 * 2 * sizeof(float));
  float* stats2 = (float*)carve(16 * 2 * sizeof(float));
  float* outpre = (float*)carve((size_t)B_ * COUT * HW2 * sizeof(float));  // ~33.5MB

  // 0) pack weights into WMMA fragment order (bf16)
  pack_frag_kernel<<<(16*4*512 + 255)/256, 256, 0, stream>>>(w1, w1p, CIN, 4, 16*4*512);
  pack_frag_kernel<<<(8*8*512  + 255)/256, 256, 0, stream>>>(w2, w2p, 2*CIN, 8, 8*8*512);
  pack_frag_kernel<<<(16*28*512+ 255)/256, 256, 0, stream>>>(weight, wcp, KDIM, 28, 16*28*512);

  // 1) GN1 stats
  const int chunk1 = (CG1 * HW1 + 63) / 64;
  gn_partial_kernel<<<16 * 64, 256, 0, stream>>>(x, part1, CIN, CG1, HW1, chunk1);
  gn_final_kernel<<<16, 64, 0, stream>>>(part1, stats1, 1.0f / (float)(CG1 * HW1));

  // 2) fused MLP + residual -> skip  (N-tile = 32)
  const int ntiles = (TOTP + 31) / 32;   // 4073
  mlp_kernel<<<ntiles, 256, 0, stream>>>(x, w1p, b1, w2p, b2, g1, be1, stats1, skip);

  // 3) DISCO conv -> outpre  (N-tile = 32)
  disco_kernel<<<B_ * HOUT * WT32, 256, 0, stream>>>(skip, psi_vals, psi_hi, psi_wi,
                                                     wcp, bias, outpre);

  // 4) GN2 stats + apply + GELU -> xdown
  const int chunk2 = (CG2 * HW2 + 63) / 64;
  gn_partial_kernel<<<16 * 64, 256, 0, stream>>>(outpre, part2, COUT, CG2, HW2, chunk2);
  gn_final_kernel<<<16, 64, 0, stream>>>(part2, stats2, 1.0f / (float)(CG2 * HW2));
  const long total2 = (long)B_ * COUT * HW2;
  gn2_apply_kernel<<<(int)((total2 + 255) / 256), 256, 0, stream>>>(outpre, stats2, g2, be2,
                                                                    xdown, total2);
}
